// Transformer_43447889166729
// MI455X (gfx1250) — compile-verified, matrix operand order checked by reading, not run on hardware
//
#include <hip/hip_runtime.h>
#include <math.h>

typedef __attribute__((ext_vector_type(16))) _Float16 v16h;
typedef __attribute__((ext_vector_type(8)))  _Float16 v8h;
typedef __attribute__((ext_vector_type(8)))  float    v8f;

#define Bsz 32
#define Cc  128
#define Ee  128
#define Hh  4
#define HSs 32
#define FFf 512
#define Ll  4
#define Vv  50000
#define Mrows (Bsz * Cc)   // 4096

__device__ __forceinline__ v8f wmma_f16(v16h a, v16h b, v8f c) {
    // (neg_a, A, neg_b, B, c_mod, C, reuse_a, reuse_b)
    return __builtin_amdgcn_wmma_f32_16x16x32_f16(false, a, false, b, (short)0, c,
                                                  false, false);
}

// Load one 16-half WMMA operand fragment for lane (tl, th):
// halves 0..7  = mem[8*th .. 8*th+7], halves 8..15 = mem[16+8*th .. 16+8*th+7]
// (matches ISA 7.12.2 16-bit A/B layout when source is contiguous in K)
__device__ __forceinline__ v16h frag(const _Float16* __restrict__ base, int th) {
    v8h lo = *(const v8h*)(base + 8 * th);
    v8h hi = *(const v8h*)(base + 16 + 8 * th);
    return __builtin_shufflevector(lo, hi, 0, 1, 2, 3, 4, 5, 6, 7,
                                   8, 9, 10, 11, 12, 13, 14, 15);
}

// ---------------------------------------------------------------- weight cvt+transpose
// out(batch, N, K) f16  <-  in(batch, K, N) f32
__global__ void k_tcvt(const float* __restrict__ in, _Float16* __restrict__ out,
                       int K, int N) {
    const float* src = in  + (size_t)blockIdx.y * K * N;
    _Float16*    dst = out + (size_t)blockIdx.y * K * N;
    int i = blockIdx.x * 256 + threadIdx.x;
    if (i < K * N) {
        int k = i / N, n = i % N;
        dst[n * K + k] = (_Float16)src[i];
    }
}

// ---------------------------------------------------------------- embedding
__global__ void k_embed(const int* __restrict__ idx,
                        const float* __restrict__ tok,
                        const float* __restrict__ pos,
                        float* __restrict__ x) {
    int row = blockIdx.x;
    int e   = threadIdx.x;
    int t   = idx[row];
    x[row * Ee + e] = tok[t * Ee + e] + pos[t * Ee + e];  // pos indexed by token id (ref bug)
}

// ---------------------------------------------------------------- layernorm (f32 in, f16 out)
__global__ void k_ln(const float* __restrict__ in,
                     const float* __restrict__ g,
                     const float* __restrict__ b,
                     _Float16* __restrict__ out, int ddof) {
    __shared__ float s[Ee];
    int row = blockIdx.x, tid = threadIdx.x;
    float v = in[row * Ee + tid];
    s[tid] = v;
    __syncthreads();
    for (int off = 64; off > 0; off >>= 1) {
        if (tid < off) s[tid] += s[tid + off];
        __syncthreads();
    }
    float mean = s[0] / (float)Ee;
    __syncthreads();
    float d = v - mean;
    s[tid] = d * d;
    __syncthreads();
    for (int off = 64; off > 0; off >>= 1) {
        if (tid < off) s[tid] += s[tid + off];
        __syncthreads();
    }
    float var = s[0] / (float)(Ee - ddof);
    out[row * Ee + tid] = (_Float16)(d * rsqrtf(var + 1e-5f) * g[tid] + b[tid]);
}

// ---------------------------------------------------------------- generic GEMM
// out(MxN) = A(MxK,f16) @ BT(NxK,f16)^T [+bias] [+resid] [relu]
// one wave computes a 32x16 tile (2 accumulators share the B fragment)
template <bool BIASF, bool RESF, bool RELUF, bool OUT16>
__global__ void k_gemm(const _Float16* __restrict__ A,
                       const _Float16* __restrict__ BT,
                       const float* __restrict__ bias,
                       const float* __restrict__ resid,
                       void* __restrict__ outv,
                       int N, int K) {
    int lane = threadIdx.x, tl = lane & 15, th = lane >> 4;
    int col0 = blockIdx.x * 16;
    int row0 = blockIdx.y * 32;
    const _Float16* Ar0  = A + (size_t)(row0 + tl) * K;
    const _Float16* Ar1  = Ar0 + (size_t)16 * K;
    const _Float16* Bcol = BT + (size_t)(col0 + tl) * K;
    v8f acc[2] = {};
    for (int k0 = 0; k0 < K; k0 += 32) {
        v16h bfrag = frag(Bcol + k0, th);
        acc[0] = wmma_f16(frag(Ar0 + k0, th), bfrag, acc[0]);
        acc[1] = wmma_f16(frag(Ar1 + k0, th), bfrag, acc[1]);
    }
    int   col  = col0 + tl;
    float bval = BIASF ? bias[col] : 0.0f;
#pragma unroll
    for (int half = 0; half < 2; ++half)
#pragma unroll
        for (int r = 0; r < 8; ++r) {
            int row = row0 + 16 * half + r + 8 * th;
            float v = acc[half][r] + bval;
            if (RESF)  v += resid[(size_t)row * N + col];
            if (RELUF) v = fmaxf(v, 0.0f);
            if (OUT16) ((_Float16*)outv)[(size_t)row * N + col] = (_Float16)v;
            else       ((float*)outv)[(size_t)row * N + col] = v;
        }
}

// ---------------------------------------------------------------- K / V projection
// acc = hbuf16(M,E) @ WT(H,HS,E)^T + bias(H,HS)
// TOUT=false: out k16 (B,H,C,HS);  TOUT=true: out vT16 (B,H,HS,C)
template <bool TOUT>
__global__ void k_gemm_kv(const _Float16* __restrict__ A,
                          const _Float16* __restrict__ WT,
                          const float* __restrict__ bias,
                          _Float16* __restrict__ out) {
    int lane = threadIdx.x, tl = lane & 15, th = lane >> 4;
    int h    = blockIdx.x >> 1;
    int col0 = (blockIdx.x & 1) * 16;
    int row0 = blockIdx.y * 32;
    const _Float16* Ar0  = A + (size_t)(row0 + tl) * Ee;
    const _Float16* Ar1  = Ar0 + (size_t)16 * Ee;
    const _Float16* Bcol = WT + (size_t)(h * HSs + col0 + tl) * Ee;
    v8f acc[2] = {};
    for (int k0 = 0; k0 < Ee; k0 += 32) {
        v16h bfrag = frag(Bcol + k0, th);
        acc[0] = wmma_f16(frag(Ar0 + k0, th), bfrag, acc[0]);
        acc[1] = wmma_f16(frag(Ar1 + k0, th), bfrag, acc[1]);
    }
    int   col  = col0 + tl;
    float bval = bias[h * HSs + col];
#pragma unroll
    for (int half = 0; half < 2; ++half)
#pragma unroll
        for (int r = 0; r < 8; ++r) {
            int row = row0 + 16 * half + r + 8 * th;   // b*C + c
            int b_  = row >> 7;
            int c_  = row & (Cc - 1);
            float v = acc[half][r] + bval;
            if (TOUT) out[((b_ * Hh + h) * HSs + col) * Cc + c_] = (_Float16)v; // v^T
            else      out[((b_ * Hh + h) * Cc + c_) * HSs + col] = (_Float16)v; // k
        }
}

// ---------------------------------------------------------------- scores + softmax
// p16(b,h,row,:) = softmax(mask(k @ k^T / sqrt(C)))   (q = k bug preserved)
// one wave: 16 rows x 128 cols, softmax in registers via half-wave shfl_xor
__global__ void k_scores(const _Float16* __restrict__ k16,
                         _Float16* __restrict__ p16) {
    int lane = threadIdx.x, tl = lane & 15, th = lane >> 4;
    int bh = blockIdx.x;
    int rb = blockIdx.y;
    const _Float16* km = k16 + (size_t)bh * Cc * HSs;

    // A fragment: rows rb*16+tl of k, K = HS = 32 (single wmma per tile)
    v16h a = frag(km + (rb * 16 + tl) * HSs, th);
    v8f acc[8];
#pragma unroll
    for (int t = 0; t < 8; ++t) {
        // B = k^T  =>  BT = k itself: fragment from row (t*16+tl)
        v16h bm = frag(km + (t * 16 + tl) * HSs, th);
        v8f z = {};
        acc[t] = wmma_f16(a, bm, z);
    }

    const float scale = 0.08838834764831843f;   // 1/sqrt(C)
#pragma unroll
    for (int t = 0; t < 8; ++t)
#pragma unroll
        for (int r = 0; r < 8; ++r) {
            int row = rb * 16 + r + 8 * th;
            int col = t * 16 + tl;
            acc[t][r] = (col <= row) ? acc[t][r] * scale : -INFINITY;
        }

#pragma unroll
    for (int r = 0; r < 8; ++r) {
        int row = rb * 16 + r + 8 * th;
        float m = -INFINITY;
#pragma unroll
        for (int t = 0; t < 8; ++t) m = fmaxf(m, acc[t][r]);
        for (int off = 1; off < 16; off <<= 1) m = fmaxf(m, __shfl_xor(m, off, 32));
        float s = 0.0f;
#pragma unroll
        for (int t = 0; t < 8; ++t) {
            float e = __expf(acc[t][r] - m);
            acc[t][r] = e;
            s += e;
        }
        for (int off = 1; off < 16; off <<= 1) s += __shfl_xor(s, off, 32);
        float inv = 1.0f / s;
#pragma unroll
        for (int t = 0; t < 8; ++t)
            p16[(size_t)bh * Cc * Cc + row * Cc + t * 16 + tl] =
                (_Float16)(acc[t][r] * inv);
    }
}

// ---------------------------------------------------------------- o = p @ v -> (B,C,E) f16
__global__ void k_attnout(const _Float16* __restrict__ p16,
                          const _Float16* __restrict__ vT16,
                          _Float16* __restrict__ o16) {
    int lane = threadIdx.x, tl = lane & 15, th = lane >> 4;
    int bh   = blockIdx.z;
    int col0 = blockIdx.x * 16;   // 0 / 16
    int row0 = blockIdx.y * 32;   // 0..96
    const _Float16* pm   = p16 + (size_t)bh * Cc * Cc;
    const _Float16* Bcol = vT16 + (size_t)(bh * HSs + col0 + tl) * Cc;
    const _Float16* Ar0  = pm + (size_t)(row0 + tl) * Cc;
    const _Float16* Ar1  = Ar0 + (size_t)16 * Cc;
    v8f acc[2] = {};
    for (int k0 = 0; k0 < Cc; k0 += 32) {
        v16h bfrag = frag(Bcol + k0, th);
        acc[0] = wmma_f16(frag(Ar0 + k0, th), bfrag, acc[0]);
        acc[1] = wmma_f16(frag(Ar1 + k0, th), bfrag, acc[1]);
    }
    int b_ = bh / Hh, h_ = bh % Hh;
    int e  = h_ * HSs + col0 + tl;
#pragma unroll
    for (int half = 0; half < 2; ++half)
#pragma unroll
        for (int r = 0; r < 8; ++r) {
            int c_ = row0 + 16 * half + r + 8 * th;
            o16[(size_t)(b_ * Cc + c_) * Ee + e] = (_Float16)acc[half][r];
        }
}

// ---------------------------------------------------------------- host
extern "C" void kernel_launch(void* const* d_in, const int* in_sizes, int n_in,
                              void* d_out, int out_size, void* d_ws, size_t ws_size,
                              hipStream_t stream) {
    const int*   idx  = (const int*)d_in[0];
    const float* tok  = (const float*)d_in[1];
    const float* pos  = (const float*)d_in[2];
    const float* Wk   = (const float*)d_in[3];
    const float* bk   = (const float*)d_in[4];
    const float* Wv   = (const float*)d_in[5];
    const float* bv   = (const float*)d_in[6];
    const float* Wo   = (const float*)d_in[7];
    const float* bo   = (const float*)d_in[8];
    const float* W1   = (const float*)d_in[9];
    const float* b1   = (const float*)d_in[10];
    const float* W2   = (const float*)d_in[11];
    const float* b2   = (const float*)d_in[12];
    const float* ln1g = (const float*)d_in[13];
    const float* ln1b = (const float*)d_in[14];
    const float* ln2g = (const float*)d_in[15];
    const float* ln2b = (const float*)d_in[16];
    const float* lnfg = (const float*)d_in[17];
    const float* lnfb = (const float*)d_in[18];
    const float* Wf   = (const float*)d_in[19];
    const float* bf   = (const float*)d_in[20];
    float* outp = (float*)d_out;

    // bump allocator over d_ws, 256B aligned
    char* base = (char*)d_ws;
    auto alloc = [&](size_t bytes) -> void* {
        void* p = (void*)base;
        base += (bytes + 255) & ~(size_t)255;
        return p;
    };
    float*     x    = (float*)alloc((size_t)Mrows * Ee * 4);
    float*     xalt = (float*)alloc((size_t)Mrows * Ee * 4);
    _Float16*  h16  = (_Float16*)alloc((size_t)Mrows * Ee * 2);
    _Float16*  k16  = (_Float16*)alloc((size_t)Bsz * Hh * Cc * HSs * 2);
    _Float16*  vT16 = (_Float16*)alloc((size_t)Bsz * Hh * HSs * Cc * 2);
    _Float16*  o16  = (_Float16*)alloc((size_t)Mrows * Ee * 2);
    _Float16*  p16  = (_Float16*)alloc((size_t)Bsz * Hh * Cc * Cc * 2);
    _Float16*  f16h = (_Float16*)alloc((size_t)Mrows * FFf * 2);
    _Float16*  wkT  = (_Float16*)alloc((size_t)Ll * Hh * Ee * HSs * 2);
    _Float16*  wvT  = (_Float16*)alloc((size_t)Ll * Hh * Ee * HSs * 2);
    _Float16*  woT  = (_Float16*)alloc((size_t)Ll * Ee * Ee * 2);
    _Float16*  w1T  = (_Float16*)alloc((size_t)Ll * Ee * FFf * 2);
    _Float16*  w2T  = (_Float16*)alloc((size_t)Ll * FFf * Ee * 2);
    _Float16*  wfT  = (_Float16*)alloc((size_t)Ee * Vv * 2);

    // -------- weight convert + transpose (deterministic, every call)
    k_tcvt<<<dim3((Ee * HSs + 255) / 256, Ll * Hh), 256, 0, stream>>>(Wk, wkT, Ee, HSs);
    k_tcvt<<<dim3((Ee * HSs + 255) / 256, Ll * Hh), 256, 0, stream>>>(Wv, wvT, Ee, HSs);
    k_tcvt<<<dim3((Ee * Ee + 255) / 256, Ll), 256, 0, stream>>>(Wo, woT, Ee, Ee);
    k_tcvt<<<dim3((Ee * FFf + 255) / 256, Ll), 256, 0, stream>>>(W1, w1T, Ee, FFf);
    k_tcvt<<<dim3((FFf * Ee + 255) / 256, Ll), 256, 0, stream>>>(W2, w2T, FFf, Ee);
    k_tcvt<<<dim3((Ee * Vv + 255) / 256, 1), 256, 0, stream>>>(Wf, wfT, Ee, Vv);

    dim3 wave(32);

    k_embed<<<Mrows, 128, 0, stream>>>(idx, tok, pos, x);

    for (int l = 0; l < Ll; ++l) {
        const _Float16* wkT_l = wkT + (size_t)l * Hh * Ee * HSs;
        const _Float16* wvT_l = wvT + (size_t)l * Hh * Ee * HSs;
        const _Float16* woT_l = woT + (size_t)l * Ee * Ee;
        const _Float16* w1T_l = w1T + (size_t)l * Ee * FFf;
        const _Float16* w2T_l = w2T + (size_t)l * FFf * Ee;
        const float* bk_l = bk + l * Hh * HSs;
        const float* bv_l = bv + l * Hh * HSs;
        const float* bo_l = bo + l * Ee;
        const float* b1_l = b1 + l * FFf;
        const float* b2_l = b2 + l * Ee;

        k_ln<<<Mrows, 128, 0, stream>>>(x, ln1g + l * Ee, ln1b + l * Ee, h16, 1);

        k_gemm_kv<false><<<dim3(Hh * 2, Mrows / 32), wave, 0, stream>>>(h16, wkT_l, bk_l, k16);
        k_gemm_kv<true ><<<dim3(Hh * 2, Mrows / 32), wave, 0, stream>>>(h16, wvT_l, bv_l, vT16);

        k_scores<<<dim3(Bsz * Hh, Cc / 16), wave, 0, stream>>>(k16, p16);
        k_attnout<<<dim3(2, Cc / 32, Bsz * Hh), wave, 0, stream>>>(p16, vT16, o16);

        // xalt = x + o @ Wo + bo
        k_gemm<true, true, false, false>
            <<<dim3(Ee / 16, Mrows / 32), wave, 0, stream>>>(o16, woT_l, bo_l, x, xalt,
                                                             Ee, Ee);
        k_ln<<<Mrows, 128, 0, stream>>>(xalt, ln2g + l * Ee, ln2b + l * Ee, h16, 1);
        // f16h = relu(h2 @ W1 + b1)   (f16 out)
        k_gemm<true, false, true, true>
            <<<dim3(FFf / 16, Mrows / 32), wave, 0, stream>>>(h16, w1T_l, b1_l, nullptr,
                                                              f16h, FFf, Ee);
        // x = xalt + f16h @ W2 + b2
        k_gemm<true, true, false, false>
            <<<dim3(Ee / 16, Mrows / 32), wave, 0, stream>>>(f16h, w2T_l, b2_l, xalt, x,
                                                             Ee, FFf);
    }

    k_ln<<<Mrows, 128, 0, stream>>>(x, lnfg, lnfb, h16, 0);
    // logits = lnf(x) @ Wf + bf  (4096 x 50000, output-write bound: ~819 MB)
    k_gemm<true, false, false, false>
        <<<dim3(Vv / 16, Mrows / 32), wave, 0, stream>>>(h16, wfT, bf, nullptr, outp,
                                                         Vv, Ee);
}
